// TemporalMamba_90082644066575
// MI455X (gfx1250) — compile-verified, matrix operand order checked by reading, not run on hardware
//
#include <hip/hip_runtime.h>
#include <hip/hip_bf16.h>
#include <math.h>

#define D_IN     1280
#define D_MODEL  512
#define N_LAYER  4
#define D_INNER  1024
#define D_STATE  16
#define D_CONV   4
#define DT_RANK  32
#define BATCH    8
#define SEQ      512
#define M_ROWS   (BATCH * SEQ)            // 4096
#define DBC_W    (DT_RANK + 2 * D_STATE)  // 64

typedef __attribute__((ext_vector_type(16))) __bf16 bf16x16;
typedef __attribute__((ext_vector_type(8)))  float  f32x8;

union BFPack { bf16x16 v; unsigned int u[8]; };

__device__ __forceinline__ unsigned int pack_bf2(float lo, float hi) {
  unsigned int a = __float_as_uint(lo);
  unsigned int b = __float_as_uint(hi);
  a += 0x7FFFu + ((a >> 16) & 1u);   // round-to-nearest-even
  b += 0x7FFFu + ((b >> 16) & 1u);
  return (a >> 16) | (b & 0xFFFF0000u);
}

__device__ __forceinline__ unsigned short bf16_bits(float f) {
  unsigned int u = __float_as_uint(f);
  u += 0x7FFFu + ((u >> 16) & 1u);
  return (unsigned short)(u >> 16);
}

// Lane's 16 bf16 operand elements: K = kh*8 + e (e<8) and 16 + kh*8 + (e-8).
// p points at row (lane&15), element column (kh*8). Two b128 loads.
__device__ __forceinline__ void load_bf_tile(const unsigned short* p, BFPack& o) {
  *reinterpret_cast<uint4*>(&o.u[0]) = *reinterpret_cast<const uint4*>(p);
  *reinterpret_cast<uint4*>(&o.u[4]) = *reinterpret_cast<const uint4*>(p + 16);
}

#define EP_STORE          0
#define EP_BIAS           1
#define EP_SOFTPLUS_BIAS  2
#define EP_ACCUM          3

// C[M,N] = A[M,K] @ W[N,K]^T, bf16 operands, f32 accumulate/output.
// Block = 128 thr = 4 waves stacked in M; wave tile = 32(M) x 64(N):
// 8 WMMAs per 12 b128 loads per K-step. grid = (N/64, M/128).
__global__ __launch_bounds__(128)
void wmma_gemm_bf16(const unsigned short* __restrict__ A,
                    const unsigned short* __restrict__ W,
                    float* __restrict__ C, unsigned short* __restrict__ Cbf,
                    const float* __restrict__ bias,
                    int K, int lda, int ldb, int ldc, int mode)
{
  const int lane = threadIdx.x & 31;
  const int wave = threadIdx.x >> 5;
  const int r    = lane & 15;
  const int kh   = lane >> 4;
  const int mTile = blockIdx.y * 128 + wave * 32;
  const int nBase = blockIdx.x * 64;

  f32x8 acc[2][4] = {};
  const unsigned short* aRow0 = A + (size_t)(mTile + r) * lda + kh * 8;
  const unsigned short* aRow1 = aRow0 + (size_t)16 * lda;
  const unsigned short* bRow  = W + (size_t)(nBase + r) * ldb + kh * 8;
  const size_t bStep = (size_t)16 * ldb;

  for (int kk = 0; kk < K; kk += 32) {
    BFPack a0, a1, b0, b1, b2, b3;
    load_bf_tile(aRow0 + kk, a0);
    load_bf_tile(aRow1 + kk, a1);
    load_bf_tile(bRow + kk,             b0);
    load_bf_tile(bRow + kk + bStep,     b1);
    load_bf_tile(bRow + kk + 2 * bStep, b2);
    load_bf_tile(bRow + kk + 3 * bStep, b3);
    acc[0][0] = __builtin_amdgcn_wmma_f32_16x16x32_bf16(false, a0.v, false, b0.v, (short)0, acc[0][0], false, false);
    acc[0][1] = __builtin_amdgcn_wmma_f32_16x16x32_bf16(false, a0.v, false, b1.v, (short)0, acc[0][1], false, false);
    acc[0][2] = __builtin_amdgcn_wmma_f32_16x16x32_bf16(false, a0.v, false, b2.v, (short)0, acc[0][2], false, false);
    acc[0][3] = __builtin_amdgcn_wmma_f32_16x16x32_bf16(false, a0.v, false, b3.v, (short)0, acc[0][3], false, false);
    acc[1][0] = __builtin_amdgcn_wmma_f32_16x16x32_bf16(false, a1.v, false, b0.v, (short)0, acc[1][0], false, false);
    acc[1][1] = __builtin_amdgcn_wmma_f32_16x16x32_bf16(false, a1.v, false, b1.v, (short)0, acc[1][1], false, false);
    acc[1][2] = __builtin_amdgcn_wmma_f32_16x16x32_bf16(false, a1.v, false, b2.v, (short)0, acc[1][2], false, false);
    acc[1][3] = __builtin_amdgcn_wmma_f32_16x16x32_bf16(false, a1.v, false, b3.v, (short)0, acc[1][3], false, false);
  }

  // D layout: lanes 0-15 -> M = v, lanes 16-31 -> M = 8+v; N = lane&15.
#pragma unroll
  for (int g = 0; g < 2; ++g) {
    const int mBase = mTile + g * 16 + kh * 8;
#pragma unroll
    for (int t = 0; t < 4; ++t) {
      const int n = nBase + t * 16 + r;
      const float bv = (mode == EP_BIAS || mode == EP_SOFTPLUS_BIAS) ? bias[n] : 0.0f;
#pragma unroll
      for (int v = 0; v < 8; ++v) {
        size_t idx = (size_t)(mBase + v) * ldc + n;
        float val = acc[g][t][v];
        if (mode == EP_BIAS) {
          val += bv;
        } else if (mode == EP_SOFTPLUS_BIAS) {
          val += bv;
          val = (val > 20.0f) ? val : log1pf(expf(val));
        } else if (mode == EP_ACCUM) {
          val += C[idx];
        }
        C[idx] = val;
        if (Cbf) Cbf[idx] = bf16_bits(val);
      }
    }
  }
}

// Bulk f32 -> bf16 conversion, 8 elems/thread (one uint4 store).
__global__ __launch_bounds__(256)
void f32_to_bf16_kernel(const float* __restrict__ in, unsigned short* __restrict__ out)
{
  const size_t i = ((size_t)blockIdx.x * 256 + threadIdx.x) * 8;
  float4 f0 = *reinterpret_cast<const float4*>(in + i);
  float4 f1 = *reinterpret_cast<const float4*>(in + i + 4);
  uint4 o;
  o.x = pack_bf2(f0.x, f0.y);
  o.y = pack_bf2(f0.z, f0.w);
  o.z = pack_bf2(f1.x, f1.y);
  o.w = pack_bf2(f1.z, f1.w);
  *reinterpret_cast<uint4*>(out + i) = o;
}

// RMSNorm over rows of 512, bf16 output (feeds a GEMM). One block/row.
__global__ __launch_bounds__(128)
void rmsnorm_bf16_kernel(const float* __restrict__ x, const float* __restrict__ w,
                         unsigned short* __restrict__ out)
{
  const int row = blockIdx.x;
  const float* xr = x + (size_t)row * D_MODEL;
  float4 v = *reinterpret_cast<const float4*>(xr + threadIdx.x * 4);
  float s = v.x * v.x + v.y * v.y + v.z * v.z + v.w * v.w;
  __shared__ float red[128];
  red[threadIdx.x] = s;
  __syncthreads();
  for (int off = 64; off > 0; off >>= 1) {
    if (threadIdx.x < off) red[threadIdx.x] += red[threadIdx.x + off];
    __syncthreads();
  }
  const float scale = rsqrtf(red[0] / (float)D_MODEL + 1e-5f);
  float4 wv = *reinterpret_cast<const float4*>(w + threadIdx.x * 4);
  uint2 o;
  o.x = pack_bf2(v.x * scale * wv.x, v.y * scale * wv.y);
  o.y = pack_bf2(v.z * scale * wv.z, v.w * scale * wv.w);
  *reinterpret_cast<uint2*>(out + (size_t)row * D_MODEL + threadIdx.x * 4) = o;
}

// RMSNorm over rows of 512, f32 output (feeds the final reduction).
__global__ __launch_bounds__(128)
void rmsnorm_f32_kernel(const float* __restrict__ x, const float* __restrict__ w,
                        float* __restrict__ out)
{
  const int row = blockIdx.x;
  const float* xr = x + (size_t)row * D_MODEL;
  float4 v = *reinterpret_cast<const float4*>(xr + threadIdx.x * 4);
  float s = v.x * v.x + v.y * v.y + v.z * v.z + v.w * v.w;
  __shared__ float red[128];
  red[threadIdx.x] = s;
  __syncthreads();
  for (int off = 64; off > 0; off >>= 1) {
    if (threadIdx.x < off) red[threadIdx.x] += red[threadIdx.x + off];
    __syncthreads();
  }
  const float scale = rsqrtf(red[0] / (float)D_MODEL + 1e-5f);
  float4 wv = *reinterpret_cast<const float4*>(w + threadIdx.x * 4);
  float4 o;
  o.x = v.x * scale * wv.x;
  o.y = v.y * scale * wv.y;
  o.z = v.z * scale * wv.z;
  o.w = v.w * scale * wv.w;
  *reinterpret_cast<float4*>(out + (size_t)row * D_MODEL + threadIdx.x * 4) = o;
}

// Depthwise causal conv (taps=4) + bias + SiLU; dual write f32 + bf16.
__global__ __launch_bounds__(256)
void conv_silu_kernel(const float* __restrict__ xr, const float* __restrict__ w,
                      const float* __restrict__ cb, float* __restrict__ xs,
                      unsigned short* __restrict__ xs_bf)
{
  const int idx = blockIdx.x * 256 + threadIdx.x;     // (b,l,d) flat
  const int d = idx & (D_INNER - 1);
  const int m = idx >> 10;
  const int l = m & (SEQ - 1);
  const int b = m >> 9;
  float acc = cb[d];
#pragma unroll
  for (int j = 0; j < D_CONV; ++j) {
    const int ll = l - (D_CONV - 1) + j;
    if (ll >= 0)
      acc += w[d * D_CONV + j] * xr[((size_t)(b * SEQ + ll)) * (2 * D_INNER) + d];
  }
  const float val = acc / (1.0f + expf(-acc));
  xs[(size_t)m * D_INNER + d] = val;
  xs_bf[(size_t)m * D_INNER + d] = bf16_bits(val);
}

// Selective scan. 16 lanes per (b,d) pair; lane = state index n.
// Fuses + xs*Dp and * silu(res); writes bf16 (consumed only by out-GEMM).
__global__ __launch_bounds__(128)
void scan_kernel(const float* __restrict__ delta, const float* __restrict__ xs,
                 const float* __restrict__ dbc, const float* __restrict__ xr,
                 const float* __restrict__ A_log, const float* __restrict__ Dp,
                 unsigned short* __restrict__ y_bf)
{
  const int lane = threadIdx.x & 31;
  const int wave = (blockIdx.x * (blockDim.x >> 5)) + (threadIdx.x >> 5);
  const int pair = wave * 2 + (lane >> 4);    // (b,d) pair, 0..8191
  const int n = lane & 15;
  const int b = pair >> 10;
  const int d = pair & (D_INNER - 1);

  const float a = -expf(A_log[d * D_STATE + n]);
  const float dcoef = Dp[d];
  float h = 0.0f;

  for (int l = 0; l < SEQ; ++l) {
    const size_t base = (size_t)(b * SEQ + l);
    const float dl = delta[base * D_INNER + d];
    const float u  = xs[base * D_INNER + d];
    const float Bv = dbc[base * DBC_W + DT_RANK + n];
    const float Cv = dbc[base * DBC_W + DT_RANK + D_STATE + n];
    h = expf(dl * a) * h + dl * Bv * u;
    float v = h * Cv;
    v += __shfl_xor(v, 8, 16);
    v += __shfl_xor(v, 4, 16);
    v += __shfl_xor(v, 2, 16);
    v += __shfl_xor(v, 1, 16);
    if (n == 0) {
      const float res = xr[base * (2 * D_INNER) + D_INNER + d];
      float out = v + u * dcoef;
      out *= res / (1.0f + expf(-res));
      y_bf[base * D_INNER + d] = bf16_bits(out);
    }
  }
}

// Per (b, channel): mean over L and unbiased std over L. Block = 512 thr.
__global__ __launch_bounds__(512)
void finalize_kernel(const float* __restrict__ m, float* __restrict__ out)
{
  const int b = blockIdx.x;
  const int c = threadIdx.x;
  float s = 0.0f, ss = 0.0f;
  for (int l = 0; l < SEQ; ++l) {
    const float v = m[((size_t)(b * SEQ + l)) * D_MODEL + c];
    s += v;
    ss += v * v;
  }
  const float mean = s / (float)SEQ;
  const float var = (ss - (float)SEQ * mean * mean) / (float)(SEQ - 1);
  out[(size_t)b * (2 * D_MODEL) + c] = mean;
  out[(size_t)b * (2 * D_MODEL) + D_MODEL + c] = sqrtf(fmaxf(var, 0.0f));
}

extern "C" void kernel_launch(void* const* d_in, const int* in_sizes, int n_in,
                              void* d_out, int out_size, void* d_ws, size_t ws_size,
                              hipStream_t stream) {
  (void)in_sizes; (void)n_in; (void)out_size; (void)ws_size;
  const float* x            = (const float*)d_in[0];
  const float* proj_w       = (const float*)d_in[1];
  const float* proj_b       = (const float*)d_in[2];
  const float* norm_w       = (const float*)d_in[3];
  const float* in_w         = (const float*)d_in[4];
  const float* conv_w       = (const float*)d_in[5];
  const float* conv_b       = (const float*)d_in[6];
  const float* xproj_w      = (const float*)d_in[7];
  const float* dt_w         = (const float*)d_in[8];
  const float* dt_b         = (const float*)d_in[9];
  const float* A_log        = (const float*)d_in[10];
  const float* Dp           = (const float*)d_in[11];
  const float* out_w        = (const float*)d_in[12];
  const float* final_norm_w = (const float*)d_in[13];
  float* out = (float*)d_out;

  // ---- workspace layout (all sizes multiples of 8 elems -> 16B aligned) ----
  const size_t SZ_X   = (size_t)M_ROWS * D_IN;          // 5,242,880
  const size_t SZ_HM  = (size_t)M_ROWS * D_MODEL;       // 2,097,152
  const size_t SZ_XR  = (size_t)M_ROWS * 2 * D_INNER;   // 8,388,608
  const size_t SZ_XS  = (size_t)M_ROWS * D_INNER;       // 4,194,304
  const size_t SZ_DBC = (size_t)M_ROWS * DBC_W;         //   262,144
  const size_t SZ_WP  = (size_t)D_MODEL * D_IN;
  const size_t SZ_WI  = (size_t)N_LAYER * 2 * D_INNER * D_MODEL;
  const size_t SZ_WX  = (size_t)N_LAYER * DBC_W * D_INNER;
  const size_t SZ_WD  = (size_t)N_LAYER * D_INNER * DT_RANK;
  const size_t SZ_WO  = (size_t)N_LAYER * D_MODEL * D_INNER;

  float* ws    = (float*)d_ws;
  float* h     = ws;
  float* xr    = h     + SZ_HM;
  float* xs    = xr    + SZ_XR;
  float* dbc   = xs    + SZ_XS;
  float* delta = dbc   + SZ_DBC;
  float* xnf   = delta + SZ_XS;
  unsigned short* x_bf      = (unsigned short*)(xnf + SZ_HM);
  unsigned short* xn_bf     = x_bf      + SZ_X;
  unsigned short* xs_bf     = xn_bf     + SZ_HM;
  unsigned short* dbc_bf    = xs_bf     + SZ_XS;
  unsigned short* y_bf      = dbc_bf    + SZ_DBC;
  unsigned short* wproj_bf  = y_bf      + SZ_XS;
  unsigned short* win_bf    = wproj_bf  + SZ_WP;
  unsigned short* wxproj_bf = win_bf    + SZ_WI;
  unsigned short* wdt_bf    = wxproj_bf + SZ_WX;
  unsigned short* wout_bf   = wdt_bf    + SZ_WD;

  // ---- one-time (per launch) bf16 staging of input + weights ----
  f32_to_bf16_kernel<<<SZ_X  / 2048, 256, 0, stream>>>(x, x_bf);
  f32_to_bf16_kernel<<<SZ_WP / 2048, 256, 0, stream>>>(proj_w, wproj_bf);
  f32_to_bf16_kernel<<<SZ_WI / 2048, 256, 0, stream>>>(in_w, win_bf);
  f32_to_bf16_kernel<<<SZ_WX / 2048, 256, 0, stream>>>(xproj_w, wxproj_bf);
  f32_to_bf16_kernel<<<SZ_WD / 2048, 256, 0, stream>>>(dt_w, wdt_bf);
  f32_to_bf16_kernel<<<SZ_WO / 2048, 256, 0, stream>>>(out_w, wout_bf);

  const dim3 blk(128);
  const int mBlk = M_ROWS / 128;   // 32

  // h = x @ proj_w.T + proj_b
  wmma_gemm_bf16<<<dim3(D_MODEL / 64, mBlk), blk, 0, stream>>>(
      x_bf, wproj_bf, h, nullptr, proj_b, D_IN, D_IN, D_IN, D_MODEL, EP_BIAS);

  for (int i = 0; i < N_LAYER; ++i) {
    rmsnorm_bf16_kernel<<<M_ROWS, 128, 0, stream>>>(h, norm_w + (size_t)i * D_MODEL, xn_bf);

    // xr = xn @ in_w[i].T  -> [xs | res]
    wmma_gemm_bf16<<<dim3(2 * D_INNER / 64, mBlk), blk, 0, stream>>>(
        xn_bf, win_bf + (size_t)i * 2 * D_INNER * D_MODEL, xr, nullptr, nullptr,
        D_MODEL, D_MODEL, D_MODEL, 2 * D_INNER, EP_STORE);

    conv_silu_kernel<<<(M_ROWS * D_INNER) / 256, 256, 0, stream>>>(
        xr, conv_w + (size_t)i * D_INNER * D_CONV, conv_b + (size_t)i * D_INNER,
        xs, xs_bf);

    // dbc = xs @ xproj_w[i].T  (f32 for scan, bf16 for dt-GEMM)
    wmma_gemm_bf16<<<dim3(DBC_W / 64, mBlk), blk, 0, stream>>>(
        xs_bf, wxproj_bf + (size_t)i * DBC_W * D_INNER, dbc, dbc_bf, nullptr,
        D_INNER, D_INNER, D_INNER, DBC_W, EP_STORE);

    // delta = softplus(dbc[:, :32] @ dt_w[i].T + dt_b[i])
    wmma_gemm_bf16<<<dim3(D_INNER / 64, mBlk), blk, 0, stream>>>(
        dbc_bf, wdt_bf + (size_t)i * D_INNER * DT_RANK, delta, nullptr,
        dt_b + (size_t)i * D_INNER, DT_RANK, DBC_W, DT_RANK, D_INNER, EP_SOFTPLUS_BIAS);

    scan_kernel<<<(BATCH * D_INNER * D_STATE) / 128, 128, 0, stream>>>(
        delta, xs, dbc, xr, A_log + (size_t)i * D_INNER * D_STATE,
        Dp + (size_t)i * D_INNER, y_bf);

    // h += y @ out_w[i].T
    wmma_gemm_bf16<<<dim3(D_MODEL / 64, mBlk), blk, 0, stream>>>(
        y_bf, wout_bf + (size_t)i * D_MODEL * D_INNER, h, nullptr, nullptr,
        D_INNER, D_INNER, D_INNER, D_MODEL, EP_ACCUM);
  }

  rmsnorm_f32_kernel<<<M_ROWS, 128, 0, stream>>>(h, final_norm_w, xnf);
  finalize_kernel<<<BATCH, D_MODEL, 0, stream>>>(xnf, out);
}
